// GATNN_attpool_14654428414343
// MI455X (gfx1250) — matrix operand order repaired and checked by reading, not compiled, not run on hardware
//
#include <hip/hip_runtime.h>
#include <hip/hip_bf16.h>

// ---------------------------------------------------------------------------
// GATv2 attention-pooling network for MI455X (gfx1250, wave32, WMMA).
// Dense GEMMs: v_wmma_f32_16x16x32_f16 with NT-way N-tile register blocking
// (A fragment loaded once, reused for NT WMMAs -> ~2.5 b128 loads / wmma).
// Irregular graph ops use VALU + global float atomics (ordered-uint max).
// ---------------------------------------------------------------------------

typedef __attribute__((ext_vector_type(16))) _Float16 v16h;
typedef __attribute__((ext_vector_type(8)))  _Float16 v8h;
typedef __attribute__((ext_vector_type(8)))  float    v8f;

#define MODE_BIAS    0   // out = acc + bias (bias may be null)
#define MODE_BN_RELU 1   // out = relu((acc+bias)*gamma*rsqrt(var+eps)+shift)
#define MODE_LEAKY   2   // out = leaky_relu(acc+bias, 0.01)

// ------------------------- WMMA GEMM -----------------------------------
// A : [M, K]   f16 row-major (K multiple of 32, M multiple of 16)
// Bt : [Npad,K] f16 row-major (== B transposed, zero-padded rows)
// D : [M, ldd] f32, columns [0,Nvalid) written. (Npad/16) % NT == 0.
template<int NT>
__global__ __launch_bounds__(256) void wmma_gemm_f16f32(
    const _Float16* __restrict__ A, const _Float16* __restrict__ Bt,
    float* __restrict__ D, int M, int Npad, int K, int ldd, int Nvalid,
    int mode, const float* __restrict__ bias,
    const float* __restrict__ bnBeta, const float* __restrict__ bnGamma,
    const float* __restrict__ bnMean, const float* __restrict__ bnVar)
{
    const int nTiles  = Npad >> 4;
    const int nGroups = nTiles / NT;
    const int mTiles  = M >> 4;
    const int wave    = blockIdx.x * (blockDim.x >> 5) + (threadIdx.x >> 5);
    if (wave >= mTiles * nGroups) return;         // wave-uniform
    const int mTile  = wave / nGroups;
    const int nTile0 = (wave % nGroups) * NT;

    const int lane = threadIdx.x & 31;
    const int row  = lane & 15;
    const int hi   = lane >> 4;                   // 0 or 1

    const _Float16* __restrict__ arow = A + (size_t)(mTile * 16 + row) * K;
    const _Float16* brow[NT];
    #pragma unroll
    for (int j = 0; j < NT; ++j)
        brow[j] = Bt + (size_t)((nTile0 + j) * 16 + row) * K + hi * 16;

    v8f acc[NT];
    #pragma unroll
    for (int j = 0; j < NT; ++j) acc[j] = (v8f){0.f,0.f,0.f,0.f,0.f,0.f,0.f,0.f};

    for (int k0 = 0; k0 < K; k0 += 32) {
        // A fragment: K = k0 + hi*16 + e (contiguous, matches B)
        v8h a0 = *(const v8h*)(arow + k0 + hi * 16);
        v8h a1 = *(const v8h*)(arow + k0 + hi * 16 + 8);
        v16h a;
        #pragma unroll
        for (int i = 0; i < 8; ++i) { a[i] = a0[i]; a[i + 8] = a1[i]; }
        // B fragments: lane column n, K = k0 + hi*16 .. +15 (contiguous)
        #pragma unroll
        for (int j = 0; j < NT; ++j) {
            v16h b = *(const v16h*)(brow[j] + k0);
            acc[j] = __builtin_amdgcn_wmma_f32_16x16x32_f16(
                         false, a, false, b, (short)0, acc[j], false, false);
        }
    }

    #pragma unroll
    for (int j = 0; j < NT; ++j) {
        const int col = (nTile0 + j) * 16 + row;
        if (col < Nvalid) {
            float bs = bias ? bias[col] : 0.f;
            float sc = 1.f, sh = 0.f;
            if (mode == MODE_BN_RELU) {
                float g = bnGamma[col] * rsqrtf(bnVar[col] + 1e-5f);
                sc = g; sh = bnBeta[col] - bnMean[col] * g;
            }
            #pragma unroll
            for (int r = 0; r < 8; ++r) {
                int m = mTile * 16 + r + hi * 8;
                float v = acc[j][r] + bs;
                if (mode == MODE_BN_RELU)    { v = v * sc + sh; v = v > 0.f ? v : 0.f; }
                else if (mode == MODE_LEAKY) { v = v > 0.f ? v : 0.01f * v; }
                D[(size_t)m * ldd + col] = v;
            }
        }
    }
}

// ------------------------- conversion / fill ---------------------------
__global__ void k_f32_to_f16_pad(const float* __restrict__ s, _Float16* __restrict__ d,
                                 int M, int K, int Kpad)
{
    long long i = (long long)blockIdx.x * blockDim.x + threadIdx.x;
    if (i >= (long long)M * Kpad) return;
    int k = (int)(i % Kpad);
    long long r = i / Kpad;
    d[i] = (k < K) ? (_Float16)s[r * K + k] : (_Float16)0.f;
}

__global__ void k_w_to_bt_f16(const float* __restrict__ W /*[K,N]*/,
                              _Float16* __restrict__ Bt /*[Npad,Kpad]*/,
                              int K, int N, int Kpad, int Npad)
{
    long long i = (long long)blockIdx.x * blockDim.x + threadIdx.x;
    if (i >= (long long)Npad * Kpad) return;
    int k = (int)(i % Kpad);
    int n = (int)(i / Kpad);
    Bt[i] = (n < N && k < K) ? (_Float16)W[(size_t)k * N + n] : (_Float16)0.f;
}

__global__ void k_fill_f32(float* p, float v, long long n)
{ long long i = (long long)blockIdx.x * blockDim.x + threadIdx.x; if (i < n) p[i] = v; }

__global__ void k_fill_u32(unsigned* p, unsigned v, long long n)
{ long long i = (long long)blockIdx.x * blockDim.x + threadIdx.x; if (i < n) p[i] = v; }

__global__ void k_bias_rows(float* __restrict__ out, const float* __restrict__ bias,
                            long long rows, int cols)
{
    long long i = (long long)blockIdx.x * blockDim.x + threadIdx.x;
    if (i >= rows * cols) return;
    out[i] = bias[i % cols];
}

// ------------------------- graph / softmax kernels ---------------------
__device__ __forceinline__ unsigned fenc(float f) {
    unsigned u = __float_as_uint(f);
    return (u & 0x80000000u) ? ~u : (u | 0x80000000u);   // order-preserving
}
__device__ __forceinline__ float fdec(unsigned u) {
    return __uint_as_float((u & 0x80000000u) ? (u & 0x7fffffffu) : ~u);
}

// alpha_logit[e,h] = sum_c leaky(xl[src]+xr[dst]+e, 0.2) * att[h,c] ; seg max
__global__ void k_edge_logits(const float* __restrict__ xl, const float* __restrict__ xr,
                              const float* __restrict__ ebuf, const float* __restrict__ att,
                              const long long* __restrict__ src, const long long* __restrict__ dst,
                              float* __restrict__ alpha, unsigned* __restrict__ nmax,
                              int E, int H, int C)
{
    long long t = (long long)blockIdx.x * blockDim.x + threadIdx.x;
    if (t >= (long long)E * H) return;
    int e = (int)(t / H), h = (int)(t % H);
    int HC = H * C;
    long long s = src[e], d = dst[e];
    const float* pl = xl   + (size_t)s * HC + h * C;
    const float* pr = xr   + (size_t)d * HC + h * C;
    const float* pe = ebuf + (size_t)e * HC + h * C;
    const float* pa = att  + h * C;
    float sum = 0.f;
    for (int c = 0; c < C; ++c) {
        float m = pl[c] + pr[c] + pe[c];
        m = m > 0.f ? m : 0.2f * m;
        sum += m * pa[c];
    }
    alpha[t] = sum;
    atomicMax(nmax + (size_t)d * H + h, fenc(sum));
}

__global__ void k_edge_exp(float* __restrict__ alpha, const long long* __restrict__ dst,
                           const unsigned* __restrict__ nmax, float* __restrict__ nsum,
                           int E, int H)
{
    long long t = (long long)blockIdx.x * blockDim.x + threadIdx.x;
    if (t >= (long long)E * H) return;
    int e = (int)(t / H), h = (int)(t % H);
    long long d = dst[e];
    float ex = __expf(alpha[t] - fdec(nmax[(size_t)d * H + h]));
    alpha[t] = ex;
    atomicAdd(nsum + (size_t)d * H + h, ex);
}

__global__ void k_edge_scatter(const float* __restrict__ xl, const float* __restrict__ alpha,
                               const float* __restrict__ nsum,
                               const long long* __restrict__ src, const long long* __restrict__ dst,
                               float* __restrict__ out, int E, int H, int C)
{
    long long t = (long long)blockIdx.x * blockDim.x + threadIdx.x;
    if (t >= (long long)E * H) return;
    int e = (int)(t / H), h = (int)(t % H);
    int HC = H * C;
    long long s = src[e], d = dst[e];
    float w = alpha[t] / nsum[(size_t)d * H + h];
    const float* pl = xl + (size_t)s * HC + h * C;
    float* po = out + (size_t)d * HC + h * C;
    for (int c = 0; c < C; ++c) atomicAdd(po + c, pl[c] * w);
}

// per-(graph,feature) softmax over nodes + weighted pool
__global__ void k_pool_max(const float* __restrict__ a, const long long* __restrict__ batch,
                           unsigned* __restrict__ pmax, int N, int F)
{
    long long t = (long long)blockIdx.x * blockDim.x + threadIdx.x;
    if (t >= (long long)N * F) return;
    int n = (int)(t / F), f = (int)(t % F);
    atomicMax(pmax + (size_t)batch[n] * F + f, fenc(a[t]));
}

__global__ void k_pool_exp_acc(const float* __restrict__ a, const float* __restrict__ h,
                               const long long* __restrict__ batch,
                               const unsigned* __restrict__ pmax,
                               float* __restrict__ psum, float* __restrict__ pnum,
                               int N, int F)
{
    long long t = (long long)blockIdx.x * blockDim.x + threadIdx.x;
    if (t >= (long long)N * F) return;
    int n = (int)(t / F), f = (int)(t % F);
    size_t gi = (size_t)batch[n] * F + f;
    float ex = __expf(a[t] - fdec(pmax[gi]));
    atomicAdd(psum + gi, ex);
    atomicAdd(pnum + gi, h[t] * ex);
}

__global__ void k_pool_div(const float* __restrict__ num, const float* __restrict__ den,
                           float* __restrict__ g, long long n)
{
    long long i = (long long)blockIdx.x * blockDim.x + threadIdx.x;
    if (i < n) g[i] = num[i] / den[i];
}

// ---------------------------------------------------------------------------
extern "C" void kernel_launch(void* const* d_in, const int* in_sizes, int n_in,
                              void* d_out, int out_size, void* d_ws, size_t ws_size,
                              hipStream_t stream)
{
    (void)in_sizes; (void)n_in; (void)out_size; (void)ws_size;
    const int N = 10000, E = 320000, G = 64;

    const float*      x     = (const float*)d_in[0];
    const long long*  eidx  = (const long long*)d_in[1];   // int64 [2,E]
    const float*      eattr = (const float*)d_in[2];       // [E,51]
    const long long*  batch = (const long long*)d_in[3];   // int64 [N]
    const long long*  src = eidx;
    const long long*  dst = eidx + E;
    auto F_ = [&](int i){ return (const float*)d_in[i]; };

    // params pytree flattened (JAX dict order: keys sorted):
    const float *We1=F_(4), *Wl1=F_(5), *Wr1=F_(6), *att1=F_(7), *bias1=F_(8), *bl1=F_(9), *br1=F_(10);
    const float *We2=F_(11),*Wl2=F_(12),*Wr2=F_(13),*att2=F_(14),*bias2=F_(15),*bl2=F_(16),*br2=F_(17);
    const float *m1W0=F_(18),*m1b0=F_(19),*m1be0=F_(20),*m1g0=F_(21),*m1m0=F_(22),*m1v0=F_(23);
    const float *m1W1=F_(24),*m1b1=F_(25),*m1be1=F_(26),*m1g1=F_(27),*m1m1=F_(28),*m1v1=F_(29);
    const float *m1W2=F_(30),*m1b2=F_(31),*m1be2=F_(32),*m1g2=F_(33),*m1m2=F_(34),*m1v2=F_(35);
    const float *m1W3=F_(36),*m1b3=F_(37);
    const float *Watt=F_(38),*batt=F_(39);
    const float *miW0=F_(40),*mib0=F_(41),*mibe0=F_(42),*mig0=F_(43),*mim0=F_(44),*miv0=F_(45);
    const float *miW1=F_(46),*mib1=F_(47);

    // ---- workspace bump allocator (peak ~620 MB) ----
    char* ws = (char*)d_ws; size_t off = 0;
    auto alloc = [&](size_t bytes)->void* {
        off = (off + 255) & ~(size_t)255; void* p = ws + off; off += bytes; return p;
    };
    _Float16* a16e = (_Float16*)alloc((size_t)E * 64 * 2);      // edge_attr f16 (K-pad 64)
    _Float16* a16n = (_Float16*)alloc((size_t)N * 384 * 2);     // node A staging
    _Float16* b16  = (_Float16*)alloc((size_t)2016 * 1024 * 2); // weight Bt staging
    float* ebuf  = (float*)alloc((size_t)E * 384 * 4);          // e = edge_attr @ We
    float* nbuf1 = (float*)alloc((size_t)N * 384 * 4);
    float* nbuf2 = (float*)alloc((size_t)N * 384 * 4);
    float* nbuf3 = (float*)alloc((size_t)N * 384 * 4);
    float* nbuf4 = (float*)alloc((size_t)N * 384 * 4);
    float* alpha = (float*)alloc((size_t)E * 4 * 4);
    unsigned* nmax = (unsigned*)alloc((size_t)N * 4 * 4);
    float*    nsum = (float*)alloc((size_t)N * 4 * 4);
    unsigned* pmax = (unsigned*)alloc((size_t)G * 384 * 4);
    float*    psum = (float*)alloc((size_t)G * 384 * 4);
    float*    pnum = (float*)alloc((size_t)G * 384 * 4);
    float*    gbuf = (float*)alloc((size_t)G * 384 * 4);
    float*    t1   = (float*)alloc((size_t)G * 1024 * 4);
    float*    t2   = (float*)alloc((size_t)G * 1024 * 4);

    auto nb = [](long long n){ return (unsigned)((n + 255) / 256); };
    auto convA = [&](const float* s, _Float16* d, int M, int K, int Kp){
        k_f32_to_f16_pad<<<nb((long long)M * Kp), 256, 0, stream>>>(s, d, M, K, Kp); };
    auto convBt = [&](const float* W, int K, int Nn, int Kp, int Np){
        k_w_to_bt_f16<<<nb((long long)Np * Kp), 256, 0, stream>>>(W, b16, K, Nn, Kp, Np); };
    auto gemm = [&](const _Float16* A, float* D, int M, int Np, int K, int ldd, int Nv,
                    int mode, const float* bias, const float* be = nullptr,
                    const float* ga = nullptr, const float* me = nullptr,
                    const float* va = nullptr){
        int nTiles = Np / 16;
        int NT = (nTiles % 4 == 0) ? 4 : ((nTiles % 3 == 0) ? 3 : 1);
        long long tiles = (long long)(M / 16) * (nTiles / NT);
        unsigned blocks = (unsigned)((tiles + 7) / 8);
        if (NT == 4)
            wmma_gemm_f16f32<4><<<blocks, 256, 0, stream>>>(A, b16, D, M, Np, K, ldd, Nv, mode, bias, be, ga, me, va);
        else if (NT == 3)
            wmma_gemm_f16f32<3><<<blocks, 256, 0, stream>>>(A, b16, D, M, Np, K, ldd, Nv, mode, bias, be, ga, me, va);
        else
            wmma_gemm_f16f32<1><<<blocks, 256, 0, stream>>>(A, b16, D, M, Np, K, ldd, Nv, mode, bias, be, ga, me, va);
    };

    // ---- mlp_init: [13->48 BN ReLU] -> [48->48] ----
    convA(x, a16n, N, 13, 32);
    convBt(miW0, 13, 48, 32, 48);
    gemm(a16n, nbuf1, N, 48, 32, 48, 48, MODE_BN_RELU, mib0, mibe0, mig0, mim0, miv0);
    convA(nbuf1, a16n, N, 48, 64);
    convBt(miW1, 48, 48, 64, 48);
    gemm(a16n, nbuf2, N, 48, 64, 48, 48, MODE_BIAS, mib1);          // h0 = nbuf2

    // ---- edge features to f16 once (shared by both GAT layers) ----
    convA(eattr, a16e, E, 51, 64);

    // ================= GAT layer 1 (H=4, C=48) =================
    convBt(We1, 51, 192, 64, 192);
    gemm(a16e, ebuf, E, 192, 64, 192, 192, MODE_BIAS, nullptr);     // e [E,192]
    convA(nbuf2, a16n, N, 48, 64);
    convBt(Wl1, 48, 192, 64, 192);
    gemm(a16n, nbuf3, N, 192, 64, 192, 192, MODE_BIAS, bl1);        // xl
    convBt(Wr1, 48, 192, 64, 192);
    gemm(a16n, nbuf4, N, 192, 64, 192, 192, MODE_BIAS, br1);        // xr
    k_fill_u32<<<nb((long long)N*4), 256, 0, stream>>>(nmax, 0u, (long long)N*4);
    k_fill_f32<<<nb((long long)N*4), 256, 0, stream>>>(nsum, 0.f, (long long)N*4);
    k_edge_logits<<<nb((long long)E*4), 256, 0, stream>>>(nbuf3, nbuf4, ebuf, att1, src, dst, alpha, nmax, E, 4, 48);
    k_edge_exp<<<nb((long long)E*4), 256, 0, stream>>>(alpha, dst, nmax, nsum, E, 4);
    k_bias_rows<<<nb((long long)N*192), 256, 0, stream>>>(nbuf1, bias1, N, 192);
    k_edge_scatter<<<nb((long long)E*4), 256, 0, stream>>>(nbuf3, alpha, nsum, src, dst, nbuf1, E, 4, 48); // h1 = nbuf1

    // ================= GAT layer 2 (H=4, C=96) =================
    convBt(We2, 51, 384, 64, 384);
    gemm(a16e, ebuf, E, 384, 64, 384, 384, MODE_BIAS, nullptr);     // e [E,384]
    convA(nbuf1, a16n, N, 192, 192);
    convBt(Wl2, 192, 384, 192, 384);
    gemm(a16n, nbuf2, N, 384, 192, 384, 384, MODE_BIAS, bl2);       // xl
    convBt(Wr2, 192, 384, 192, 384);
    gemm(a16n, nbuf4, N, 384, 192, 384, 384, MODE_BIAS, br2);       // xr
    k_fill_u32<<<nb((long long)N*4), 256, 0, stream>>>(nmax, 0u, (long long)N*4);
    k_fill_f32<<<nb((long long)N*4), 256, 0, stream>>>(nsum, 0.f, (long long)N*4);
    k_edge_logits<<<nb((long long)E*4), 256, 0, stream>>>(nbuf2, nbuf4, ebuf, att2, src, dst, alpha, nmax, E, 4, 96);
    k_edge_exp<<<nb((long long)E*4), 256, 0, stream>>>(alpha, dst, nmax, nsum, E, 4);
    k_bias_rows<<<nb((long long)N*384), 256, 0, stream>>>(nbuf3, bias2, N, 384);
    k_edge_scatter<<<nb((long long)E*4), 256, 0, stream>>>(nbuf2, alpha, nsum, src, dst, nbuf3, E, 4, 96); // h2 = nbuf3

    // ---- attention pooling ----
    convA(nbuf3, a16n, N, 384, 384);
    convBt(Watt, 384, 384, 384, 384);
    gemm(a16n, nbuf1, N, 384, 384, 384, 384, MODE_BIAS, batt);      // att logits
    k_fill_u32<<<nb((long long)G*384), 256, 0, stream>>>(pmax, 0u, (long long)G*384);
    k_fill_f32<<<nb((long long)G*384), 256, 0, stream>>>(psum, 0.f, (long long)G*384);
    k_fill_f32<<<nb((long long)G*384), 256, 0, stream>>>(pnum, 0.f, (long long)G*384);
    k_pool_max<<<nb((long long)N*384), 256, 0, stream>>>(nbuf1, batch, pmax, N, 384);
    k_pool_exp_acc<<<nb((long long)N*384), 256, 0, stream>>>(nbuf1, nbuf3, batch, pmax, psum, pnum, N, 384);
    k_pool_div<<<nb((long long)G*384), 256, 0, stream>>>(pnum, psum, gbuf, (long long)G*384);

    // ---- mlp1: 384->1024->1024->1024 (BN+ReLU) -> 2001 (leaky 0.01) ----
    convA(gbuf, a16n, G, 384, 384);
    convBt(m1W0, 384, 1024, 384, 1024);
    gemm(a16n, t1, G, 1024, 384, 1024, 1024, MODE_BN_RELU, m1b0, m1be0, m1g0, m1m0, m1v0);
    convA(t1, a16n, G, 1024, 1024);
    convBt(m1W1, 1024, 1024, 1024, 1024);
    gemm(a16n, t2, G, 1024, 1024, 1024, 1024, MODE_BN_RELU, m1b1, m1be1, m1g1, m1m1, m1v1);
    convA(t2, a16n, G, 1024, 1024);
    convBt(m1W2, 1024, 1024, 1024, 1024);
    gemm(a16n, t1, G, 1024, 1024, 1024, 1024, MODE_BN_RELU, m1b2, m1be2, m1g2, m1m2, m1v2);
    convA(t1, a16n, G, 1024, 1024);
    convBt(m1W3, 1024, 2001, 1024, 2016);
    gemm(a16n, (float*)d_out, G, 2016, 1024, 2001, 2001, MODE_LEAKY, m1b3);
}
